// Attention_27874337751091
// MI455X (gfx1250) — matrix operand order repaired and checked by reading, not compiled, not run on hardware
//
#include <hip/hip_runtime.h>
#include <hip/hip_bf16.h>

// ---------------------------------------------------------------------------
// Attention block for MI455X (gfx1250): GroupNorm -> QKV GEMM -> QK^T ->
// softmax -> PV -> proj(+bias)+residual.
//   * all GEMMs on matrix cores: v_wmma_f32_16x16x32_bf16
//   * B^T block tiles staged into LDS by the Tensor Data Mover with
//     DOUBLE BUFFERING: TDM fill of chunk i+1 overlaps WMMA on chunk i
//     (tensor_load_to_lds + s_wait_tensorcnt, in-order TENSORcnt semantics).
//   * TDM row padding (4 DWORDs / 256B row) for conflict-free ds_load_b128.
// ---------------------------------------------------------------------------

typedef __bf16 bf16;
typedef __attribute__((ext_vector_type(16))) __bf16 v16bf;
typedef __attribute__((ext_vector_type(8)))  __bf16 v8bf;
typedef __attribute__((ext_vector_type(8)))  float  v8f;
typedef __attribute__((ext_vector_type(4)))  unsigned int u32x4;
typedef __attribute__((ext_vector_type(8)))  int i32x8;
typedef __attribute__((ext_vector_type(4)))  int i32x4;

#define WMMA_BF16(a, b, c) \
  __builtin_amdgcn_wmma_f32_16x16x32_bf16(false, (a), false, (b), (short)0, (c), false, false)

// Problem constants (fixed by the reference).
static constexpr int kB = 16;    // batch
static constexpr int kC = 512;   // channels
static constexpr int kN = 1024;  // tokens = H*W
static constexpr int kG = 32;    // groups
static constexpr float kScale = 0.04419417382415922f;  // 512^-0.5 folded into q

// GEMM tiling: block = 8 waves (4 M x 2 N), wave tile 32x64 (2x4 WMMA tiles).
// Block tile 128(M) x 128(N).  B^T staged in LDS in K-chunks of 128 elements,
// double-buffered.
static constexpr int kChunk    = 128;                 // K elements per stage
static constexpr int kLdsLd    = kChunk + 8;          // +8 bf16 = +4 DWORD TDM pad
static constexpr int kBN       = 128;                 // B^T rows per block
static constexpr int kBufElems = kBN * kLdsLd;        // 17408 bf16 = 34816 B
static constexpr unsigned kBufBytes = kBufElems * sizeof(bf16);

// ---------------------------------------------------------------------------
// TDM: stage a [rows x k_elems] K-major tile (row stride ld_elems, bf16) into
// LDS with 4-DWORD padding after each 256-byte row (ISA D# layout, ch.8).
// ---------------------------------------------------------------------------
__device__ __forceinline__ void tdm_load_bt_chunk(const bf16* gsrc, unsigned lds_off,
                                                  int rows, int k_elems, int ld_elems) {
  unsigned long long ga = (unsigned long long)(uintptr_t)(const void*)gsrc;
  u32x4 g0;
  g0[0] = 1u;                                               // count=1 (valid user D#)
  g0[1] = lds_off;                                          // lds_addr (bytes)
  g0[2] = (unsigned)(ga & 0xFFFFFFFFu);                     // global_addr[31:0]
  g0[3] = (unsigned)((ga >> 32) & 0x1FFFFFFu) | (2u << 30); // addr[56:32] | type=2
  i32x8 g1;
  g1[0] = (1 << 16)      // data_size = 1 -> 2 bytes
        | (1 << 20)      // pad_enable
        | (5 << 22)      // pad_interval: (1<<5)*8B = 256B (one 128-elem bf16 row)
        | (3 << 25);     // pad_amount: 3+1 = 4 DWORDs
  g1[1] = (int)((k_elems & 0xFFFF) << 16);                                  // tensor_dim0 lo
  g1[2] = (int)(((k_elems >> 16) & 0xFFFF) | ((rows & 0xFFFF) << 16));      // dim0 hi | dim1 lo
  g1[3] = (int)(((rows >> 16) & 0xFFFF) | ((k_elems & 0xFFFF) << 16));      // dim1 hi | tile_dim0
  g1[4] = rows & 0xFFFF;                                                    // tile_dim1 (tile_dim2=0)
  g1[5] = ld_elems;                                                         // tensor_dim0_stride lo
  g1[6] = 0;
  g1[7] = 0;
  i32x4 z4 = {0, 0, 0, 0};
#if __clang_major__ >= 23
  i32x8 z8 = {0, 0, 0, 0, 0, 0, 0, 0};
  __builtin_amdgcn_tensor_load_to_lds(g0, g1, z4, z4, z8, 0);
#else
  __builtin_amdgcn_tensor_load_to_lds(g0, g1, z4, z4, 0);
#endif
}

// ---------------------------------------------------------------------------
// Fragment loaders matching CDNA5 wave32 WMMA VGPR layouts (ISA 7.12.2).
// ---------------------------------------------------------------------------
__device__ __forceinline__ v16bf load_a_frag(const bf16* __restrict__ a_tile,
                                             int lda, int lane) {
  int m  = lane & 15;
  int kb = (lane >> 4) * 8;
  const bf16* p = a_tile + (size_t)m * lda + kb;
  v8bf lo = *(const v8bf*)(p);
  v8bf hi = *(const v8bf*)(p + 16);
  v16bf r;
#pragma unroll
  for (int i = 0; i < 8; ++i) { r[i] = lo[i]; r[i + 8] = hi[i]; }
  return r;
}

__device__ __forceinline__ v16bf load_b_frag_lds(const bf16* bt_tile, int ld, int lane) {
  int n  = lane & 15;
  int kb = (lane >> 4) * 8;
  const bf16* p = bt_tile + n * ld + kb;
  v8bf lo = *(const v8bf*)(p);
  v8bf hi = *(const v8bf*)(p + 16);
  v16bf r;
#pragma unroll
  for (int i = 0; i < 8; ++i) { r[i] = lo[i]; r[i + 8] = hi[i]; }
  return r;
}

__device__ __forceinline__ int dfrag_row(int i, int r, int lane) {
  return i * 16 + (lane >> 4) * 8 + r;
}
__device__ __forceinline__ int dfrag_col(int j, int lane) {
  return j * 16 + (lane & 15);
}

// ---------------------------------------------------------------------------
// GEMM core: wave computes 32(M) x 64(N); block of 8 waves computes 128x128.
// A read from global (K-major rows); B^T double-buffered in LDS via TDM:
//   wave0: TDM(chunk0) ; per chunk: wait tensorcnt -> barrier ->
//          issue TDM(chunk+1, other buf) -> WMMA burst -> barrier.
// ---------------------------------------------------------------------------
template <int K, int LDA, int LDB>
__device__ __forceinline__ void gemm_core(const bf16* __restrict__ A_wave,
                                          const bf16* __restrict__ Bt_blk,
                                          bf16* smB, int lane, int wn,
                                          v8f acc[2][4]) {
  constexpr int NCH = K / kChunk;
  const unsigned lds_base = (unsigned)(uintptr_t)(void*)smB;

  if (threadIdx.x < 32) {
    tdm_load_bt_chunk(Bt_blk, lds_base, kBN, kChunk, LDB);
  }

  for (int ic = 0; ic < NCH; ++ic) {
    if (threadIdx.x < 32) __builtin_amdgcn_s_wait_tensorcnt(0);  // chunk ic landed
    __syncthreads();                                             // publish to all waves

    // Prefetch next chunk into the other LDS buffer while we compute.
    if (threadIdx.x < 32 && (ic + 1) < NCH) {
      tdm_load_bt_chunk(Bt_blk + (size_t)(ic + 1) * kChunk,
                        lds_base + ((unsigned)((ic + 1) & 1)) * kBufBytes,
                        kBN, kChunk, LDB);
    }

    const bf16* buf = smB + (size_t)(ic & 1) * kBufElems;
    int kc = ic * kChunk;
    // Warm L2/L0 for the next A K-chunk (speculative prefetch).
    if ((ic + 1) < NCH) {
      __builtin_prefetch(A_wave + (size_t)(lane & 15) * LDA + kc + kChunk, 0, 1);
    }

#pragma unroll
    for (int k0 = 0; k0 < kChunk; k0 += 32) {
      v16bf a0 = load_a_frag(A_wave + kc + k0, LDA, lane);
      v16bf a1 = load_a_frag(A_wave + (size_t)16 * LDA + kc + k0, LDA, lane);
      v16bf bfr[4];
#pragma unroll
      for (int j = 0; j < 4; ++j)
        bfr[j] = load_b_frag_lds(buf + (wn * 64 + j * 16) * kLdsLd + k0, kLdsLd, lane);
#pragma unroll
      for (int j = 0; j < 4; ++j) {
        acc[0][j] = WMMA_BF16(a0, bfr[j], acc[0][j]);
        acc[1][j] = WMMA_BF16(a1, bfr[j], acc[1][j]);
      }
    }
    __syncthreads();  // all waves done with buf before it is refilled (ic+2)
  }
}

// ---------------------------------------------------------------------------
// Kernel 0: f32 -> bf16 weight conversion.
// ---------------------------------------------------------------------------
__global__ __launch_bounds__(256) void cvt_bf16_kernel(const float* __restrict__ src,
                                                       bf16* __restrict__ dst, int n) {
  int i = blockIdx.x * 256 + threadIdx.x;
  if (i < n) dst[i] = (bf16)src[i];
}

// ---------------------------------------------------------------------------
// Kernel 1: GroupNorm.  One block per (batch, group): 16 channels x 1024
// spatial = 16384 contiguous floats.  Writes xn token-major [b][n][c] bf16.
// ---------------------------------------------------------------------------
__global__ __launch_bounds__(256) void gn_kernel(const float* __restrict__ x,
                                                 const float* __restrict__ w,
                                                 const float* __restrict__ bias,
                                                 bf16* __restrict__ xn) {
  int bg = blockIdx.x;
  int b = bg >> 5, g = bg & 31;
  const float* xp = x + (size_t)(b * kC + g * 16) * kN;
  int tid = threadIdx.x;

  float s = 0.f, s2 = 0.f;
  for (int i = tid; i < 16384; i += 256) {
    float v = xp[i];
    s += v; s2 += v * v;
  }
  __shared__ float r1[256], r2[256];
  r1[tid] = s; r2[tid] = s2;
  __syncthreads();
  for (int off = 128; off > 0; off >>= 1) {
    if (tid < off) { r1[tid] += r1[tid + off]; r2[tid] += r2[tid + off]; }
    __syncthreads();
  }
  float mean = r1[0] * (1.f / 16384.f);
  float var  = r2[0] * (1.f / 16384.f) - mean * mean;
  float rstd = rsqrtf(var + 1e-5f);

  for (int i = tid; i < 16384; i += 256) {
    int cl = i >> 10, n = i & 1023;
    int c = g * 16 + cl;
    float val = (xp[i] - mean) * rstd * w[c] + bias[c];
    xn[((size_t)b * kN + n) * kC + c] = (bf16)val;
  }
}

// ---------------------------------------------------------------------------
// Kernel 2: QKV GEMM.  D[o][n] = sum_c W[o][c] * xn[b][n][c] + bias[o].
// Grid: (1536/128, 1024/128, B) = (12, 8, 16).
// ---------------------------------------------------------------------------
__global__ __launch_bounds__(256) void qkv_gemm_kernel(const bf16* __restrict__ W,
                                                       const bf16* __restrict__ xn,
                                                       const float* __restrict__ qkv_bias,
                                                       bf16* __restrict__ qT,
                                                       bf16* __restrict__ kT,
                                                       bf16* __restrict__ vbuf) {
  __shared__ bf16 smB[2 * kBufElems];
  int lane = threadIdx.x & 31, wave = threadIdx.x >> 5;
  int wm = wave >> 1, wn = wave & 1;
  int row0 = blockIdx.x * 128 + wm * 32;   // o
  int col0 = blockIdx.y * 128;             // n (block)
  int b = blockIdx.z;

  const bf16* A  = W + (size_t)row0 * kC;
  const bf16* Bt = xn + (size_t)b * kN * kC + (size_t)col0 * kC;

  v8f acc[2][4] = {};
  gemm_core<kC, kC, kC>(A, Bt, smB, lane, wn, acc);

  size_t bo = (size_t)b * kN * kC;
#pragma unroll
  for (int i = 0; i < 2; ++i)
#pragma unroll
    for (int j = 0; j < 4; ++j)
#pragma unroll
      for (int r = 0; r < 8; ++r) {
        int row = row0 + dfrag_row(i, r, lane);            // o
        int col = col0 + wn * 64 + dfrag_col(j, lane);     // n
        float val = acc[i][j][r] + qkv_bias[row];
        if (row < kC) {
          qT[bo + (size_t)col * kC + row] = (bf16)(val * kScale);
        } else if (row < 2 * kC) {
          kT[bo + (size_t)col * kC + (row - kC)] = (bf16)val;
        } else {
          vbuf[bo + (size_t)(row - 2 * kC) * kN + col] = (bf16)val;
        }
      }
}

// ---------------------------------------------------------------------------
// Kernel 3: scores S[n][m] = sum_c qT[n][c] * kT[m][c].  Grid: (8, 8).
// ---------------------------------------------------------------------------
__global__ __launch_bounds__(256) void s_gemm_kernel(const bf16* __restrict__ qT,
                                                     const bf16* __restrict__ kT,
                                                     float* __restrict__ S, int b) {
  __shared__ bf16 smB[2 * kBufElems];
  int lane = threadIdx.x & 31, wave = threadIdx.x >> 5;
  int wm = wave >> 1, wn = wave & 1;
  int row0 = blockIdx.x * 128 + wm * 32;   // n
  int col0 = blockIdx.y * 128;             // m (block)

  const bf16* A  = qT + (size_t)b * kN * kC + (size_t)row0 * kC;
  const bf16* Bt = kT + (size_t)b * kN * kC + (size_t)col0 * kC;

  v8f acc[2][4] = {};
  gemm_core<kC, kC, kC>(A, Bt, smB, lane, wn, acc);

#pragma unroll
  for (int i = 0; i < 2; ++i)
#pragma unroll
    for (int j = 0; j < 4; ++j)
#pragma unroll
      for (int r = 0; r < 8; ++r) {
        int row = row0 + dfrag_row(i, r, lane);
        int col = col0 + wn * 64 + dfrag_col(j, lane);
        S[(size_t)row * kN + col] = acc[i][j][r];
      }
}

// ---------------------------------------------------------------------------
// Kernel 4: row softmax, f32 in -> bf16 out.  One block per row.
// ---------------------------------------------------------------------------
__global__ __launch_bounds__(256) void softmax_kernel(const float* __restrict__ S,
                                                      bf16* __restrict__ P) {
  int row = blockIdx.x, tid = threadIdx.x;
  const float* sp = S + (size_t)row * kN;

  float v[4];
  float m = -3.4e38f;
#pragma unroll
  for (int i = 0; i < 4; ++i) {
    v[i] = sp[tid + 256 * i];
    m = fmaxf(m, v[i]);
  }
  __shared__ float red[256];
  red[tid] = m;
  __syncthreads();
  for (int off = 128; off > 0; off >>= 1) {
    if (tid < off) red[tid] = fmaxf(red[tid], red[tid + off]);
    __syncthreads();
  }
  m = red[0];
  __syncthreads();

  float s = 0.f;
#pragma unroll
  for (int i = 0; i < 4; ++i) {
    v[i] = __expf(v[i] - m);
    s += v[i];
  }
  red[tid] = s;
  __syncthreads();
  for (int off = 128; off > 0; off >>= 1) {
    if (tid < off) red[tid] += red[tid + off];
    __syncthreads();
  }
  float rs = 1.f / red[0];

  bf16* pp = P + (size_t)row * kN;
#pragma unroll
  for (int i = 0; i < 4; ++i) pp[tid + 256 * i] = (bf16)(v[i] * rs);
}

// ---------------------------------------------------------------------------
// Kernel 5: O[n][c] = sum_m P[n][m] * v[c][m].  Grid: (8, 4).
// ---------------------------------------------------------------------------
__global__ __launch_bounds__(256) void pv_gemm_kernel(const bf16* __restrict__ P,
                                                      const bf16* __restrict__ vbuf,
                                                      bf16* __restrict__ oT, int b) {
  __shared__ bf16 smB[2 * kBufElems];
  int lane = threadIdx.x & 31, wave = threadIdx.x >> 5;
  int wm = wave >> 1, wn = wave & 1;
  int row0 = blockIdx.x * 128 + wm * 32;   // n
  int col0 = blockIdx.y * 128;             // c (block)

  const bf16* A  = P + (size_t)row0 * kN;
  const bf16* Bt = vbuf + (size_t)b * kC * kN + (size_t)col0 * kN;

  v8f acc[2][4] = {};
  gemm_core<kN, kN, kN>(A, Bt, smB, lane, wn, acc);

  size_t bo = (size_t)b * kN * kC;
#pragma unroll
  for (int i = 0; i < 2; ++i)
#pragma unroll
    for (int j = 0; j < 4; ++j)
#pragma unroll
      for (int r = 0; r < 8; ++r) {
        int row = row0 + dfrag_row(i, r, lane);          // n
        int col = col0 + wn * 64 + dfrag_col(j, lane);   // c
        oT[bo + (size_t)row * kC + col] = (bf16)acc[i][j][r];
      }
}

// ---------------------------------------------------------------------------
// Kernel 6: out[b][co][n] = sum_c Wp[co][c]*o[c][n] + pbias[co] + o[co][n].
// Grid: (4, 8, B).
// ---------------------------------------------------------------------------
__global__ __launch_bounds__(256) void proj_gemm_kernel(const bf16* __restrict__ Wp,
                                                        const bf16* __restrict__ oT,
                                                        const float* __restrict__ pbias,
                                                        float* __restrict__ out) {
  __shared__ bf16 smB[2 * kBufElems];
  int lane = threadIdx.x & 31, wave = threadIdx.x >> 5;
  int wm = wave >> 1, wn = wave & 1;
  int row0 = blockIdx.x * 128 + wm * 32;   // co
  int col0 = blockIdx.y * 128;             // n (block)
  int b = blockIdx.z;

  const bf16* A  = Wp + (size_t)row0 * kC;
  const bf16* Bt = oT + (size_t)b * kN * kC + (size_t)col0 * kC;

  v8f acc[2][4] = {};
  gemm_core<kC, kC, kC>(A, Bt, smB, lane, wn, acc);

  size_t bo_o = (size_t)b * kN * kC;
  size_t bo_y = (size_t)b * kC * kN;
#pragma unroll
  for (int i = 0; i < 2; ++i)
#pragma unroll
    for (int j = 0; j < 4; ++j)
#pragma unroll
      for (int r = 0; r < 8; ++r) {
        int row = row0 + dfrag_row(i, r, lane);          // co
        int col = col0 + wn * 64 + dfrag_col(j, lane);   // n
        float o_res = (float)oT[bo_o + (size_t)col * kC + row];  // o[co][n]
        out[bo_y + (size_t)row * kN + col] = acc[i][j][r] + pbias[row] + o_res;
      }
}

// ---------------------------------------------------------------------------
// Host launcher.
// ---------------------------------------------------------------------------
extern "C" void kernel_launch(void* const* d_in, const int* in_sizes, int n_in,
                              void* d_out, int out_size, void* d_ws, size_t ws_size,
                              hipStream_t stream) {
  const float* x       = (const float*)d_in[0];
  const float* gn_w    = (const float*)d_in[1];
  const float* gn_b    = (const float*)d_in[2];
  const float* qkv_w   = (const float*)d_in[3];
  const float* qkv_b   = (const float*)d_in[4];
  const float* proj_w  = (const float*)d_in[5];
  const float* proj_b  = (const float*)d_in[6];
  float* out = (float*)d_out;

  // Workspace sub-allocation (256-byte aligned).
  char* ws = (char*)d_ws;
  size_t off = 0;
  auto alloc = [&](size_t bytes) -> void* {
    void* p = ws + off;
    off += (bytes + 255) & ~(size_t)255;
    return p;
  };
  bf16* Wqkv = (bf16*)alloc((size_t)3 * kC * kC * sizeof(bf16));   // 1.5 MB
  bf16* Wp   = (bf16*)alloc((size_t)kC * kC * sizeof(bf16));       // 0.5 MB
  bf16* xn   = (bf16*)alloc((size_t)kB * kN * kC * sizeof(bf16));  // 16 MB
  bf16* qT   = (bf16*)alloc((size_t)kB * kN * kC * sizeof(bf16));  // 16 MB
  bf16* kT   = (bf16*)alloc((size_t)kB * kN * kC * sizeof(bf16));  // 16 MB
  bf16* vb   = (bf16*)alloc((size_t)kB * kC * kN * sizeof(bf16));  // 16 MB
  float* S   = (float*)alloc((size_t)kN * kN * sizeof(float));     // 4 MB (per-batch)
  bf16* P    = (bf16*)alloc((size_t)kN * kN * sizeof(bf16));       // 2 MB (per-batch)
  bf16* oT   = xn;  // xn is dead after the QKV GEMM; reuse for attention output.

  // 0) weights -> bf16
  cvt_bf16_kernel<<<(3 * kC * kC + 255) / 256, 256, 0, stream>>>(qkv_w, Wqkv, 3 * kC * kC);
  cvt_bf16_kernel<<<(kC * kC + 255) / 256, 256, 0, stream>>>(proj_w, Wp, kC * kC);

  // 1) GroupNorm -> xn [b][n][c] bf16
  gn_kernel<<<kB * kG, 256, 0, stream>>>(x, gn_w, gn_b, xn);

  // 2) QKV GEMM (scale folded into q)
  qkv_gemm_kernel<<<dim3(12, 8, kB), 256, 0, stream>>>(Wqkv, xn, qkv_b, qT, kT, vb);

  // 3) per-batch attention: scores -> softmax -> PV (S/P scratch recycled)
  for (int b = 0; b < kB; ++b) {
    s_gemm_kernel<<<dim3(8, 8), 256, 0, stream>>>(qT, kT, S, b);
    softmax_kernel<<<kN, 256, 0, stream>>>(S, P);
    pv_gemm_kernel<<<dim3(8, 4), 256, 0, stream>>>(P, vb, oT, b);
  }

  // 4) projection + bias + (replicated-bug) residual with attention output
  proj_gemm_kernel<<<dim3(4, 8, kB), 256, 0, stream>>>(Wp, oT, proj_b, out);
}
